// GraphSAGEWithJK_16045997818029
// MI455X (gfx1250) — compile-verified, matrix operand order checked by reading, not run on hardware
//
#include <hip/hip_runtime.h>
#include <hip/hip_bf16.h>
#include <math.h>

// ---------------------------------------------------------------------------
// GraphSAGE + JumpingKnowledge(cat) + mean-pool + MLP head, CDNA5 (gfx1250).
//
// N=50000 nodes, E=800000 edges, F=H=128, L=4 layers, G=128 graphs, C=10.
// Layer GEMMs fused as one K=256 bf16 WMMA GEMM:
//   out = relu([agg*inv_deg | h] @ [Wl;Wr] + bl)
// A tile is converted to packed-bf16 once in the LDS fill stage; the WMMA
// inner loop is then pure ds_load_b128 + global_load_b128 + v_wmma.
// ---------------------------------------------------------------------------

#define NNODES 50000
#define NEDGES 800000
#define HDIM   128
#define NLAYER 4
#define NGRAPH 128
#define NCLASS 10
#define KFUSED 256           // 2*HDIM
#define JKDIM  (NLAYER*HDIM) // 512
#define ASTRIDE 132          // LDS row stride in dwords (128 + 4 pad, mult of 4)

typedef __attribute__((ext_vector_type(16))) __bf16 v16bf;
typedef __attribute__((ext_vector_type(8)))  float  v8f;

union BF16x16 { v16bf v; unsigned int u[8]; };

// round-to-nearest-even f32 -> bf16, pack two into a dword
__device__ __forceinline__ unsigned int pack2_bf16(float lo, float hi) {
    unsigned int ul = __float_as_uint(lo);
    unsigned int uh = __float_as_uint(hi);
    ul = (ul + 0x7FFFu + ((ul >> 16) & 1u)) >> 16;
    uh = (uh + 0x7FFFu + ((uh >> 16) & 1u)) >> 16;
    return (ul & 0xFFFFu) | (uh << 16);
}

// K-within-32-block position for A/B 16-bit fragment (CDNA5 ISA):
//   lanes 0-15:  vgpr r<4 -> K=2r,2r+1 ; r>=4 -> K=16+2(r-4)+h
//   lanes 16-31: vgpr r<4 -> K=8+2r+h  ; r>=4 -> K=24+2(r-4)+h
__device__ __forceinline__ int klocal(int lane, int r, int h) {
    return ((lane >= 16) ? 8 : 0) + ((r >= 4) ? 16 : 0) + 2 * (r & 3) + h;
}

// ---------------------------------------------------------------------------
// 1. degree / inv_deg
// ---------------------------------------------------------------------------
__global__ __launch_bounds__(256) void k_deg(const int* __restrict__ dst,
                                             float* __restrict__ deg, int E) {
    int e = blockIdx.x * 256 + threadIdx.x;
    if (e < E) atomicAdd(&deg[dst[e]], 1.0f);
}

__global__ __launch_bounds__(256) void k_invdeg(float* __restrict__ deg, int n) {
    int i = blockIdx.x * 256 + threadIdx.x;
    if (i < n) {
        float d = deg[i];
        deg[i] = 1.0f / fmaxf(d, 1.0f);
    }
}

// ---------------------------------------------------------------------------
// 2. pack [Wl;Wr] per layer into wave32 WMMA B-fragment layout.
//    Tile id t = l*64 + kb*8 + nb ; dword index = t*256 + lane*8 + r.
//    Each lane then loads its 8 B dwords as two contiguous b128 loads.
// ---------------------------------------------------------------------------
__global__ __launch_bounds__(256) void k_pack_w(const float* __restrict__ Wl,
                                                const float* __restrict__ Wr,
                                                unsigned int* __restrict__ Wp) {
    int tid = blockIdx.x * 256 + threadIdx.x;     // [0, L*64*32*8)
    int r    = tid & 7;
    int lane = (tid >> 3) & 31;
    int t    = tid >> 8;                          // tile index, [0, 256)
    int l    = t >> 6;
    int rem  = t & 63;
    int kb   = rem >> 3;
    int nb   = rem & 7;
    int n    = nb * 16 + (lane & 15);
    float f[2];
#pragma unroll
    for (int h = 0; h < 2; ++h) {
        int k = kb * 32 + klocal(lane, r, h);     // fused K in [0,256)
        f[h] = (k < HDIM) ? Wl[(size_t)l * HDIM * HDIM + (size_t)k * HDIM + n]
                          : Wr[(size_t)l * HDIM * HDIM + (size_t)(k - HDIM) * HDIM + n];
    }
    Wp[(size_t)t * 256 + lane * 8 + r] = pack2_bf16(f[0], f[1]);
}

// ---------------------------------------------------------------------------
// 3. mean-aggregation scatter: one wave per edge, lane covers 4 features.
// ---------------------------------------------------------------------------
__global__ __launch_bounds__(256) void k_scatter(const float* __restrict__ h, int hstride,
                                                 const int* __restrict__ src,
                                                 const int* __restrict__ dst,
                                                 float* __restrict__ agg, int E) {
    int e = blockIdx.x * 8 + (threadIdx.x >> 5);
    if (e >= E) return;
    int lane = threadIdx.x & 31;
    int s = src[e], d = dst[e];
    const float4 v = *(const float4*)(h + (size_t)s * hstride + lane * 4);
    float* ap = agg + (size_t)d * HDIM + lane * 4;
    atomicAdd(ap + 0, v.x);
    atomicAdd(ap + 1, v.y);
    atomicAdd(ap + 2, v.z);
    atomicAdd(ap + 3, v.w);
}

// ---------------------------------------------------------------------------
// 4. fused SAGEConv GEMM: out[16x128 tile] = relu([agg*invdeg | h] @ Wp + b)
//    256 threads = 8 waves; wave w owns the 16-col tile nb = w.
//    LDS A tile stored as packed bf16 K-pairs so each lane's A fragment for a
//    K=32 block is exactly two contiguous ds_load_b128s (no VALU repack).
// ---------------------------------------------------------------------------
__global__ __launch_bounds__(256) void k_sage_gemm(const float* __restrict__ agg,
                                                   const float* __restrict__ invdeg,
                                                   const float* __restrict__ h, int hstride,
                                                   const unsigned int* __restrict__ Wp,
                                                   const float* __restrict__ bias,
                                                   float* __restrict__ out, int ostride) {
    __shared__ unsigned int As[16 * ASTRIDE];     // 16 rows x 128 bf16-pair dwords
    const int tid  = threadIdx.x;
    const int row0 = blockIdx.x * 16;             // 50000 = 3125*16, no tail

    // cooperative fill: fused operand row = [agg*invdeg (K 0..127) | h (K 128..255)]
    // dword dw holds K = {2dw, 2dw+1}; both on same side of the 128 boundary.
#pragma unroll
    for (int i = 0; i < 8; ++i) {
        int idx  = tid + i * 256;                 // [0, 2048)
        int r    = idx >> 7;                      // row 0..15
        int dw   = idx & 127;                     // dword 0..127
        int node = row0 + r;
        int k    = dw * 2;
        float f0, f1;
        if (k < HDIM) {
            float inv = invdeg[node];
            f0 = agg[(size_t)node * HDIM + k]     * inv;
            f1 = agg[(size_t)node * HDIM + k + 1] * inv;
        } else {
            f0 = h[(size_t)node * hstride + (k - HDIM)];
            f1 = h[(size_t)node * hstride + (k - HDIM + 1)];
        }
        As[r * ASTRIDE + dw] = pack2_bf16(f0, f1);
    }
    __syncthreads();

    const int wave = tid >> 5;                    // nb, column tile
    const int lane = tid & 31;
    const int m    = lane & 15;
    const unsigned int* arow = As + m * ASTRIDE + ((lane >= 16) ? 4 : 0);

    v8f acc = {};
#pragma unroll
    for (int kb = 0; kb < 8; ++kb) {
        // A fragment: dwords base+{0..3} and base+{8..11} of this K=32 block
        BF16x16 afrag;
        uint4 a0 = *(const uint4*)(arow + kb * 16);
        uint4 a1 = *(const uint4*)(arow + kb * 16 + 8);
        afrag.u[0] = a0.x; afrag.u[1] = a0.y; afrag.u[2] = a0.z; afrag.u[3] = a0.w;
        afrag.u[4] = a1.x; afrag.u[5] = a1.y; afrag.u[6] = a1.z; afrag.u[7] = a1.w;

        // B fragment: pre-packed, two contiguous b128 loads per lane
        BF16x16 bfrag;
        const uint4* bp = (const uint4*)(Wp + (size_t)(kb * 8 + wave) * 256 + lane * 8);
        uint4 b0 = bp[0], b1 = bp[1];
        bfrag.u[0] = b0.x; bfrag.u[1] = b0.y; bfrag.u[2] = b0.z; bfrag.u[3] = b0.w;
        bfrag.u[4] = b1.x; bfrag.u[5] = b1.y; bfrag.u[6] = b1.z; bfrag.u[7] = b1.w;

        acc = __builtin_amdgcn_wmma_f32_16x16x32_bf16(
            /*neg_a=*/false, afrag.v, /*neg_b=*/false, bfrag.v,
            /*c_mod=*/(short)0, acc, /*reuse_a=*/false, /*reuse_b=*/false);
    }

    // epilogue: + bias, ReLU, store (C layout: vgpr r -> M=r (+8 for hi lanes))
    const int col  = wave * 16 + (lane & 15);
    const float b  = bias[col];
    const int moff = (lane >= 16) ? 8 : 0;
#pragma unroll
    for (int r = 0; r < 8; ++r) {
        int node = row0 + r + moff;
        float v  = acc[r] + b;
        v = v > 0.0f ? v : 0.0f;
        out[(size_t)node * ostride + col] = v;
    }
}

// ---------------------------------------------------------------------------
// 5. global mean pool: one wave per node over the 512-dim JK concat.
// ---------------------------------------------------------------------------
__global__ __launch_bounds__(256) void k_pool(const float* __restrict__ hcat,
                                              const int* __restrict__ batch,
                                              float* __restrict__ gsum,
                                              float* __restrict__ gcnt, int n) {
    int node = blockIdx.x * 8 + (threadIdx.x >> 5);
    if (node >= n) return;
    int lane = threadIdx.x & 31;
    int b = batch[node];
#pragma unroll
    for (int j = 0; j < JKDIM / 32; ++j) {
        int c = lane + j * 32;
        atomicAdd(&gsum[(size_t)b * JKDIM + c], hcat[(size_t)node * JKDIM + c]);
    }
    if (lane == 0) atomicAdd(&gcnt[b], 1.0f);
}

// ---------------------------------------------------------------------------
// 6. head: fc1 (512->128, relu) then fc2 (128->10) + log_softmax
// ---------------------------------------------------------------------------
__global__ __launch_bounds__(128) void k_fc1(const float* __restrict__ gsum,
                                             const float* __restrict__ gcnt,
                                             const float* __restrict__ W1,
                                             const float* __restrict__ b1,
                                             float* __restrict__ gact) {
    int g = blockIdx.x;        // graph
    int o = threadIdx.x;       // output feature
    float inv = 1.0f / fmaxf(gcnt[g], 1.0f);
    float s = b1[o];
    for (int k = 0; k < JKDIM; ++k)
        s += gsum[(size_t)g * JKDIM + k] * inv * W1[(size_t)k * HDIM + o];
    gact[(size_t)g * HDIM + o] = s > 0.0f ? s : 0.0f;
}

__global__ __launch_bounds__(128) void k_fc2(const float* __restrict__ gact,
                                             const float* __restrict__ W2,
                                             const float* __restrict__ b2,
                                             float* __restrict__ out) {
    int g = threadIdx.x;       // one graph per thread (G == 128)
    float logits[NCLASS];
#pragma unroll
    for (int c = 0; c < NCLASS; ++c) logits[c] = b2[c];
    for (int k = 0; k < HDIM; ++k) {
        float a = gact[(size_t)g * HDIM + k];
#pragma unroll
        for (int c = 0; c < NCLASS; ++c) logits[c] += a * W2[(size_t)k * NCLASS + c];
    }
    float m = logits[0];
#pragma unroll
    for (int c = 1; c < NCLASS; ++c) m = fmaxf(m, logits[c]);
    float s = 0.0f;
#pragma unroll
    for (int c = 0; c < NCLASS; ++c) s += __expf(logits[c] - m);
    float lse = m + __logf(s);
#pragma unroll
    for (int c = 0; c < NCLASS; ++c) out[(size_t)g * NCLASS + c] = logits[c] - lse;
}

// ---------------------------------------------------------------------------
// launch
// ---------------------------------------------------------------------------
extern "C" void kernel_launch(void* const* d_in, const int* in_sizes, int n_in,
                              void* d_out, int out_size, void* d_ws, size_t ws_size,
                              hipStream_t stream) {
    (void)in_sizes; (void)n_in; (void)out_size; (void)ws_size;

    const float* x    = (const float*)d_in[0];              // [N,128]
    const int*   eidx = (const int*)d_in[1];                // [2,E]
    const int*   batch= (const int*)d_in[2];                // [N]
    const float* Wl   = (const float*)d_in[3];              // [L,128,128]
    const float* bl   = (const float*)d_in[4];              // [L,128]
    const float* Wr   = (const float*)d_in[5];              // [L,128,128]
    const float* W1   = (const float*)d_in[6];              // [512,128]
    const float* b1   = (const float*)d_in[7];              // [128]
    const float* W2   = (const float*)d_in[8];              // [128,10]
    const float* b2   = (const float*)d_in[9];              // [10]
    float* out = (float*)d_out;                             // [128,10]

    const int* src = eidx;
    const int* dst = eidx + NEDGES;

    // workspace carve-up (256-byte aligned slices)
    char* p = (char*)d_ws;
    auto take = [&](size_t bytes) -> char* {
        char* q = p;
        p += (bytes + 255) & ~(size_t)255;
        return q;
    };
    float*        invdeg = (float*)take((size_t)NNODES * 4);                 // deg -> inv_deg
    float*        agg    = (float*)take((size_t)NNODES * HDIM * 4);          // 25.6 MB
    float*        hcat   = (float*)take((size_t)NNODES * JKDIM * 4);         // 102.4 MB
    unsigned int* Wp     = (unsigned int*)take((size_t)NLAYER * 64 * 256 * 4);
    float*        gsum   = (float*)take((size_t)NGRAPH * JKDIM * 4);
    float*        gcnt   = (float*)take((size_t)NGRAPH * 4);
    float*        gact   = (float*)take((size_t)NGRAPH * HDIM * 4);

    // degrees
    hipMemsetAsync(invdeg, 0, (size_t)NNODES * 4, stream);
    k_deg<<<(NEDGES + 255) / 256, 256, 0, stream>>>(dst, invdeg, NEDGES);
    k_invdeg<<<(NNODES + 255) / 256, 256, 0, stream>>>(invdeg, NNODES);

    // weight pre-pack to WMMA B-fragment layout (L*64*32*8 = 65536 dwords)
    k_pack_w<<<256, 256, 0, stream>>>(Wl, Wr, Wp);

    // SAGE layers
    for (int l = 0; l < NLAYER; ++l) {
        const float* h  = (l == 0) ? x : (hcat + (size_t)(l - 1) * HDIM);
        const int hstr  = (l == 0) ? HDIM : JKDIM;
        hipMemsetAsync(agg, 0, (size_t)NNODES * HDIM * 4, stream);
        k_scatter<<<NEDGES / 8, 256, 0, stream>>>(h, hstr, src, dst, agg, NEDGES);
        k_sage_gemm<<<NNODES / 16, 256, 0, stream>>>(
            agg, invdeg, h, hstr,
            Wp + (size_t)l * 64 * 256,
            bl + (size_t)l * HDIM,
            hcat + (size_t)l * HDIM, JKDIM);
    }

    // pooling
    hipMemsetAsync(gsum, 0, (size_t)NGRAPH * JKDIM * 4, stream);
    hipMemsetAsync(gcnt, 0, (size_t)NGRAPH * 4, stream);
    k_pool<<<(NNODES + 7) / 8, 256, 0, stream>>>(hcat, batch, gsum, gcnt, NNODES);

    // head
    k_fc1<<<NGRAPH, 128, 0, stream>>>(gsum, gcnt, W1, b1, gact);
    k_fc2<<<1, 128, 0, stream>>>(gact, W2, b2, out);
}
